// GATv2Backbone_35201551958232
// MI455X (gfx1250) — matrix operand order repaired and checked by reading, not compiled, not run on hardware
//
#include <hip/hip_runtime.h>

typedef __attribute__((ext_vector_type(16))) __bf16 v16bf;
typedef __attribute__((ext_vector_type(8)))  __bf16 v8bf;
typedef __attribute__((ext_vector_type(8)))  float  v8f;

#define N_NODES 50000
#define E_EDGES 800000
#define ET      (E_EDGES + N_NODES)   // edges + self loops
#define IN_DIM  128
#define HC      256                   // HEADS * HIDDEN
#define MTILES  (N_NODES / 16)        // 3125 (exact)
#define NEG_SLOPE 0.2f
#define LN_EPS    1e-5f

// ---------- order-preserving float <-> uint for atomic segment-max ----------
__device__ __forceinline__ unsigned f2o(float f) {
  unsigned u = __float_as_uint(f);
  return (u & 0x80000000u) ? ~u : (u | 0x80000000u);
}
__device__ __forceinline__ float o2f(unsigned u) {
  u = (u & 0x80000000u) ? (u & 0x7fffffffu) : ~u;
  return __uint_as_float(u);
}

__device__ __forceinline__ void edge_ids(const long long* __restrict__ ei, int e,
                                         int& src, int& dst) {
  if (e < E_EDGES) { src = (int)ei[e]; dst = (int)ei[E_EDGES + e]; }
  else             { src = dst = e - E_EDGES; }   // self loops appended
}

// ------------------------------- utility fills ------------------------------
__global__ void fill_u32_k(unsigned* __restrict__ p, unsigned v, long long n) {
  long long i = (long long)blockIdx.x * blockDim.x + threadIdx.x;
  if (i < n) p[i] = v;
}

__global__ void f32_to_bf16_k(const float* __restrict__ in, __bf16* __restrict__ out,
                              long long n) {
  long long i = (long long)blockIdx.x * blockDim.x + threadIdx.x;
  if (i < n) out[i] = (__bf16)in[i];
}

// W is (K x HC) row-major f32 -> Wt is (HC x K) bf16 (N-major for WMMA B loads)
__global__ void w_transpose_bf16_k(const float* __restrict__ W, __bf16* __restrict__ Wt,
                                   int K) {
  int i = blockIdx.x * blockDim.x + threadIdx.x;
  if (i < K * HC) {
    int k = i / HC, n = i % HC;
    Wt[(size_t)n * K + k] = (__bf16)W[i];
  }
}

// --------------------------- WMMA bf16 GEMM ---------------------------------
// C[M x HC] = A[M x K](bf16) * B[K x HC](bf16, stored N-major as Bt[HC x K]) + bias
// Register-blocked: each wave computes a 32(M) x 64(N) macro-tile
//   (2 m-subtiles x 4 n-subtiles) -> A frag reused 4x, B frag reused 2x.
// Block = 4 waves covering all 16 n-tiles; grid.x covers ceil(3125/2) m-pairs.
__global__ void __launch_bounds__(128)
gemm_wmma_bf16_k(const __bf16* __restrict__ A, const __bf16* __restrict__ Bt,
                 const float* __restrict__ bias, float* __restrict__ C, int K) {
  const int wave = threadIdx.x >> 5;
  const int lane = threadIdx.x & 31;
  const int g = lane >> 4;     // half-wave group
  const int r = lane & 15;

  const int nt0 = wave * 4;                       // 4 n-subtiles per wave
  const int mt0 = blockIdx.x * 2;                 // 2 m-subtiles per wave
  const bool m1_valid = (mt0 + 1) < MTILES;
  const int mt1 = m1_valid ? (mt0 + 1) : (MTILES - 1);  // clamp loads, guard stores

  const __bf16* Arow0 = A + (size_t)(mt0 * 16 + r) * K;
  const __bf16* Arow1 = A + (size_t)(mt1 * 16 + r) * K;
  const __bf16* Brow0 = Bt + (size_t)(nt0 * 16 + r) * K;   // n-subtiles are +16 rows apart

  v8f acc[2][4] = {};

  for (int k0 = 0; k0 < K; k0 += 32) {
    // A 16-bit 16x32 layout: lanes 0-15 hold K 0..7 / 16..23, lanes 16-31 +8
    v8bf a0lo = *reinterpret_cast<const v8bf*>(Arow0 + k0 + 8 * g);
    v8bf a0hi = *reinterpret_cast<const v8bf*>(Arow0 + k0 + 16 + 8 * g);
    v8bf a1lo = *reinterpret_cast<const v8bf*>(Arow1 + k0 + 8 * g);
    v8bf a1hi = *reinterpret_cast<const v8bf*>(Arow1 + k0 + 16 + 8 * g);
    v16bf a0, a1;
#pragma unroll
    for (int i = 0; i < 8; i++) {
      a0[i] = a0lo[i]; a0[i + 8] = a0hi[i];
      a1[i] = a1lo[i]; a1[i + 8] = a1hi[i];
    }
#pragma unroll
    for (int n = 0; n < 4; n++) {
      // B 32x16: lanes 0-15 hold K 0..7 / 16..23, lanes 16-31 +8 (same as A)
      v8bf blo = *reinterpret_cast<const v8bf*>(Brow0 + (size_t)n * 16 * K + k0 + 8 * g);
      v8bf bhi = *reinterpret_cast<const v8bf*>(Brow0 + (size_t)n * 16 * K + k0 + 16 + 8 * g);
      v16bf b = __builtin_shufflevector(blo, bhi, 0, 1, 2, 3, 4, 5, 6, 7, 8, 9, 10, 11, 12, 13, 14, 15);
      acc[0][n] = __builtin_amdgcn_wmma_f32_16x16x32_bf16(false, a0, false, b,
                                                          (short)0, acc[0][n], false, false);
      acc[1][n] = __builtin_amdgcn_wmma_f32_16x16x32_bf16(false, a1, false, b,
                                                          (short)0, acc[1][n], false, false);
    }
  }

#pragma unroll
  for (int n = 0; n < 4; n++) {
    const int col = (nt0 + n) * 16 + r;
    const float bc = bias[col];
#pragma unroll
    for (int v = 0; v < 8; v++) {
      const int m0 = mt0 * 16 + v + 8 * g;         // C layout: row = v + 8*g
      C[(size_t)m0 * HC + col] = acc[0][n][v] + bc;
      if (m1_valid) {
        const int m1 = mt1 * 16 + v + 8 * g;
        C[(size_t)m1 * HC + col] = acc[1][n][v] + bc;
      }
    }
  }
}

// ----------------------- per-edge attention logits ---------------------------
// one wave per edge; lane covers 8 channels (all within one head)
__global__ void __launch_bounds__(256)
edge_logits_k(const float* __restrict__ xl, const float* __restrict__ xr,
              const float* __restrict__ att, const long long* __restrict__ ei,
              float* __restrict__ logits) {
  const int e = blockIdx.x * 8 + (threadIdx.x >> 5);
  if (e >= ET) return;
  const int lane = threadIdx.x & 31;
  int src, dst; edge_ids(ei, e, src, dst);
  const int c0 = lane * 8;
  const float4* pl = reinterpret_cast<const float4*>(xl + (size_t)src * HC + c0);
  const float4* pr = reinterpret_cast<const float4*>(xr + (size_t)dst * HC + c0);
  const float4* pa = reinterpret_cast<const float4*>(att + c0);
  float s = 0.f;
#pragma unroll
  for (int q = 0; q < 2; q++) {
    float4 l4 = pl[q], r4 = pr[q], a4 = pa[q];
    float v;
    v = l4.x + r4.x; s += (v > 0.f ? v : NEG_SLOPE * v) * a4.x;
    v = l4.y + r4.y; s += (v > 0.f ? v : NEG_SLOPE * v) * a4.y;
    v = l4.z + r4.z; s += (v > 0.f ? v : NEG_SLOPE * v) * a4.z;
    v = l4.w + r4.w; s += (v > 0.f ? v : NEG_SLOPE * v) * a4.w;
  }
#pragma unroll
  for (int m = 1; m < 8; m <<= 1) s += __shfl_xor(s, m, 32);  // reduce 8-lane head group
  if ((lane & 7) == 0) logits[(size_t)e * 4 + (lane >> 3)] = s;
}

// ------------------------- segment softmax passes ----------------------------
__global__ void seg_max_k(const float* __restrict__ logits,
                          const long long* __restrict__ ei,
                          unsigned* __restrict__ maxo) {
  int i = blockIdx.x * blockDim.x + threadIdx.x;
  if (i >= ET * 4) return;
  int e = i >> 2, h = i & 3;
  int src, dst; edge_ids(ei, e, src, dst); (void)src;
  atomicMax(&maxo[dst * 4 + h], f2o(logits[i]));
}

__global__ void seg_expsum_k(float* __restrict__ logits,
                             const long long* __restrict__ ei,
                             const unsigned* __restrict__ maxo,
                             float* __restrict__ denom) {
  int i = blockIdx.x * blockDim.x + threadIdx.x;
  if (i >= ET * 4) return;
  int e = i >> 2, h = i & 3;
  int src, dst; edge_ids(ei, e, src, dst); (void)src;
  float ex = __expf(logits[i] - o2f(maxo[dst * 4 + h]));
  logits[i] = ex;                                   // overwrite with exp value
  atomicAdd(&denom[dst * 4 + h], ex);
}

// ------------------------- weighted scatter-add ------------------------------
__global__ void __launch_bounds__(256)
aggregate_k(const float* __restrict__ xl, const float* __restrict__ ex,
            const float* __restrict__ denom, const long long* __restrict__ ei,
            float* __restrict__ agg) {
  const int e = blockIdx.x * 8 + (threadIdx.x >> 5);
  if (e >= ET) return;
  const int lane = threadIdx.x & 31;
  int src, dst; edge_ids(ei, e, src, dst);
  const int head = lane >> 3;
  const float alpha = ex[(size_t)e * 4 + head] / denom[dst * 4 + head];
  const int c0 = lane * 8;
  const float4* pl = reinterpret_cast<const float4*>(xl + (size_t)src * HC + c0);
  float* po = agg + (size_t)dst * HC + c0;
#pragma unroll
  for (int q = 0; q < 2; q++) {
    float4 v = pl[q];
    atomicAdd(po + q * 4 + 0, alpha * v.x);
    atomicAdd(po + q * 4 + 1, alpha * v.y);
    atomicAdd(po + q * 4 + 2, alpha * v.z);
    atomicAdd(po + q * 4 + 3, alpha * v.w);
  }
}

// ------------------ bias + ReLU + LayerNorm + requantize ---------------------
__global__ void __launch_bounds__(256)
epilogue_k(const float* __restrict__ agg, const float* __restrict__ bias,
           const float* __restrict__ lng, const float* __restrict__ lnb,
           __bf16* __restrict__ hbf, float* __restrict__ fout) {
  const int n = blockIdx.x, c = threadIdx.x;
  float v = agg[(size_t)n * HC + c] + bias[c];
  v = fmaxf(v, 0.f);                                   // ReLU
  __shared__ float s1[HC], s2[HC];
  s1[c] = v; s2[c] = v * v;
  __syncthreads();
  for (int off = 128; off > 0; off >>= 1) {
    if (c < off) { s1[c] += s1[c + off]; s2[c] += s2[c + off]; }
    __syncthreads();
  }
  const float mu  = s1[0] * (1.f / HC);
  const float var = s2[0] * (1.f / HC) - mu * mu;
  const float rs  = rsqrtf(var + LN_EPS);
  const float o = (v - mu) * rs * lng[c] + lnb[c];
  hbf[(size_t)n * HC + c] = (__bf16)o;                 // next layer GEMM input
  if (fout) fout[(size_t)n * HC + c] = o;              // final f32 output
}

// ---------------------------------------------------------------------------
extern "C" void kernel_launch(void* const* d_in, const int* in_sizes, int n_in,
                              void* d_out, int out_size, void* d_ws, size_t ws_size,
                              hipStream_t stream) {
  (void)in_sizes; (void)n_in; (void)out_size; (void)ws_size;
  const float*     x  = (const float*)d_in[0];
  const long long* ei = (const long long*)d_in[1];

  char* w = (char*)d_ws;
  auto alloc = [&](size_t bytes) -> void* {
    void* p = (void*)w; w += (bytes + 255) & ~(size_t)255; return p;
  };
  __bf16*   hbf   = (__bf16*)  alloc((size_t)N_NODES * HC * 2);   // 25.6 MB
  __bf16*   WtL   = (__bf16*)  alloc((size_t)HC * HC * 2);
  __bf16*   WtR   = (__bf16*)  alloc((size_t)HC * HC * 2);
  float*    xl    = (float*)   alloc((size_t)N_NODES * HC * 4);   // 51.2 MB
  float*    xragg = (float*)   alloc((size_t)N_NODES * HC * 4);   // xr, reused as agg
  float*    logit = (float*)   alloc((size_t)ET * 4 * 4);         // 13.6 MB
  unsigned* maxo  = (unsigned*)alloc((size_t)N_NODES * 4 * 4);
  float*    denom = (float*)   alloc((size_t)N_NODES * 4 * 4);

  // layer-0 input -> bf16
  {
    long long n = (long long)N_NODES * IN_DIM;
    f32_to_bf16_k<<<(unsigned)((n + 255) / 256), 256, 0, stream>>>(x, hbf, n);
  }

  int K = IN_DIM;
  for (int l = 0; l < 3; l++) {
    const float* Wl   = (const float*)d_in[2 + 8 * l + 0];
    const float* bl   = (const float*)d_in[2 + 8 * l + 1];
    const float* Wr   = (const float*)d_in[2 + 8 * l + 2];
    const float* br   = (const float*)d_in[2 + 8 * l + 3];
    const float* att  = (const float*)d_in[2 + 8 * l + 4];
    const float* bias = (const float*)d_in[2 + 8 * l + 5];
    const float* lng  = (const float*)d_in[2 + 8 * l + 6];
    const float* lnb  = (const float*)d_in[2 + 8 * l + 7];

    const unsigned tb = (unsigned)((K * HC + 255) / 256);
    w_transpose_bf16_k<<<tb, 256, 0, stream>>>(Wl, WtL, K);
    w_transpose_bf16_k<<<tb, 256, 0, stream>>>(Wr, WtR, K);

    // 4 waves/block cover all 16 n-tiles; each wave does 2 m-tiles x 4 n-tiles
    const unsigned gmx = (MTILES + 1) / 2;        // 1563
    gemm_wmma_bf16_k<<<dim3(gmx, 1), 128, 0, stream>>>(hbf, WtL, bl, xl, K);
    gemm_wmma_bf16_k<<<dim3(gmx, 1), 128, 0, stream>>>(hbf, WtR, br, xragg, K);

    const unsigned eb = (ET + 7) / 8;             // 1 wave per edge, 8 waves/block
    edge_logits_k<<<eb, 256, 0, stream>>>(xl, xragg, att, ei, logit);

    const unsigned nb4 = (N_NODES * 4 + 255) / 256;
    fill_u32_k<<<nb4, 256, 0, stream>>>(maxo, 0u, (long long)N_NODES * 4);
    fill_u32_k<<<nb4, 256, 0, stream>>>((unsigned*)denom, 0u, (long long)N_NODES * 4);

    const unsigned ebh = (ET * 4 + 255) / 256;
    seg_max_k   <<<ebh, 256, 0, stream>>>(logit, ei, maxo);
    seg_expsum_k<<<ebh, 256, 0, stream>>>(logit, ei, maxo, denom);

    const long long aggn = (long long)N_NODES * HC;
    fill_u32_k<<<(unsigned)((aggn + 255) / 256), 256, 0, stream>>>((unsigned*)xragg, 0u, aggn);
    aggregate_k<<<eb, 256, 0, stream>>>(xl, logit, denom, ei, xragg);

    float* fout = (l == 2) ? (float*)d_out : nullptr;
    epilogue_k<<<N_NODES, HC, 0, stream>>>(xragg, bias, lng, lnb, hbf, fout);
    K = HC;
  }
}
